// MixLoss_10333691314719
// MI455X (gfx1250) — compile-verified, matrix operand order checked by reading, not run on hardware
//
#include <hip/hip_runtime.h>
#include <hip/hip_bf16.h>
#include <math.h>

#define BATCH 512
#define FEAT  512
#define NCLS  100000
#define NTILE 32
#define KCHUNKS (FEAT / 32)
#define WB_STRIDE 528   // 512 + 16 pad; row stride = 1056 B (multiple of 32B, offsets banks)

typedef __attribute__((ext_vector_type(16))) __bf16        v16bf;
typedef __attribute__((ext_vector_type(8)))  float          v8f;
typedef __attribute__((ext_vector_type(16))) unsigned short v16us;

__device__ __forceinline__ unsigned short f2bf(float x) {
    unsigned u = __builtin_bit_cast(unsigned, x);
    u += 0x7FFFu + ((u >> 16) & 1u);          // round-to-nearest-even
    return (unsigned short)(u >> 16);
}

// Kernel 1: L2-normalize input rows -> bf16 A [512 x 512]; zero per-row sumexp.
__global__ void __launch_bounds__(256) prep_inputs(const float* __restrict__ inp,
                                                   unsigned short* __restrict__ Abf,
                                                   float* __restrict__ rowsum) {
    int row = blockIdx.x;
    int t   = threadIdx.x;                    // 256 threads, 8 waves
    float a = inp[row * FEAT + t];
    float b = inp[row * FEAT + t + 256];
    float s = a * a + b * b;
#pragma unroll
    for (int m = 16; m >= 1; m >>= 1) s += __shfl_xor(s, m, 32);
    __shared__ float red[8];
    int wave = t >> 5, lane = t & 31;
    if (lane == 0) red[wave] = s;
    __syncthreads();
    float tot = 0.f;
#pragma unroll
    for (int i = 0; i < 8; ++i) tot += red[i];
    float inv = 1.0f / fmaxf(sqrtf(tot), 1e-12f);
    Abf[row * FEAT + t]       = f2bf(a * inv);
    Abf[row * FEAT + t + 256] = f2bf(b * inv);
    if (t == 0) rowsum[row] = 0.0f;
}

// Kernel 2: 512 threads = 16 waves (4 per SIMD -> wave-level latency hiding).
// Per block: normalize 32 weight rows -> bf16 LDS, then 512x32 GEMM tile via
// v_wmma_f32_16x16x32_bf16; each wave computes a 32x32 sub-tile (2 Mt x 2 Nt).
// Writes outs = 32*clip(cos); accumulates per-row sum of exp(out-32).
__global__ void __launch_bounds__(512) margin_gemm(const unsigned short* __restrict__ Abf,
                                                   const float* __restrict__ W,
                                                   float* __restrict__ outs,
                                                   float* __restrict__ rowsum) {
    __shared__ __attribute__((aligned(32))) unsigned short Wb[NTILE * WB_STRIDE];

    int n0   = blockIdx.x * NTILE;
    int wave = threadIdx.x >> 5;              // 16 waves
    int lane = threadIdx.x & 31;

    // --- Stage 1: normalize NTILE weight rows into LDS (each wave: 2 rows) ---
#pragma unroll
    for (int rr = 0; rr < 2; ++rr) {
        int r = wave * 2 + rr;
        const float4* wrow = (const float4*)(W + (size_t)(n0 + r) * FEAT);
        float4 v[4];
        float  s = 0.f;
#pragma unroll
        for (int j = 0; j < 4; ++j) {
            v[j] = wrow[lane + 32 * j];
            s += v[j].x * v[j].x + v[j].y * v[j].y + v[j].z * v[j].z + v[j].w * v[j].w;
        }
#pragma unroll
        for (int m = 16; m >= 1; m >>= 1) s += __shfl_xor(s, m, 32);
        float inv = 1.0f / fmaxf(sqrtf(s), 1e-12f);
#pragma unroll
        for (int j = 0; j < 4; ++j) {
            ushort4 p;
            p.x = f2bf(v[j].x * inv);
            p.y = f2bf(v[j].y * inv);
            p.z = f2bf(v[j].z * inv);
            p.w = f2bf(v[j].w * inv);
            *(ushort4*)&Wb[r * WB_STRIDE + (lane + 32 * j) * 4] = p;
        }
    }
    __syncthreads();

    // --- Stage 2: WMMA GEMM. Wave covers M rows [wave*32, wave*32+32), N = NTILE. ---
    int m0   = wave * 32;
    int half = lane >> 4;                     // K-half (operands) / M-half (C/D)
    int l15  = lane & 15;

    // Stable base pointers: every load below is base + compile-time offset,
    // so no per-chunk address VALU (and no WMMA->VALU hazard NOPs).
    const unsigned short* aPtr0 = Abf + (size_t)(m0 + l15) * FEAT + half * 16;
    const unsigned short* aPtr1 = aPtr0 + 16 * FEAT;
    const unsigned short* wPtr0 = &Wb[l15 * WB_STRIDE + half * 16];
    const unsigned short* wPtr1 = wPtr0 + 16 * WB_STRIDE;

    v8f acc[2][2] = {};                       // 2 M-tiles x 2 N-tiles of 16x16 f32

    v16bf abuf[2][2];
    v16bf bbuf[2][2];

    // Prologue: load chunk 0 operands.
    bbuf[0][0] = __builtin_bit_cast(v16bf, *(const v16us*)wPtr0);
    bbuf[0][1] = __builtin_bit_cast(v16bf, *(const v16us*)wPtr1);
    abuf[0][0] = __builtin_bit_cast(v16bf, *(const v16us*)aPtr0);
    abuf[0][1] = __builtin_bit_cast(v16bf, *(const v16us*)aPtr1);

#pragma unroll
    for (int c = 0; c < KCHUNKS; ++c) {
        int cur = c & 1, nxt = cur ^ 1;
        // Issue next chunk's loads before this chunk's WMMAs (latency hiding).
        if (c + 1 < KCHUNKS) {
            int kc = (c + 1) * 32;
            bbuf[nxt][0] = __builtin_bit_cast(v16bf, *(const v16us*)(wPtr0 + kc));
            bbuf[nxt][1] = __builtin_bit_cast(v16bf, *(const v16us*)(wPtr1 + kc));
            abuf[nxt][0] = __builtin_bit_cast(v16bf, *(const v16us*)(aPtr0 + kc));
            abuf[nxt][1] = __builtin_bit_cast(v16bf, *(const v16us*)(aPtr1 + kc));
        }
#pragma unroll
        for (int mt = 0; mt < 2; ++mt)
#pragma unroll
            for (int nt = 0; nt < 2; ++nt)
                acc[mt][nt] = __builtin_amdgcn_wmma_f32_16x16x32_bf16(
                    false, abuf[cur][mt], false, bbuf[cur][nt], (short)0,
                    acc[mt][nt], false, false);
    }

    // --- Epilogue: clip, scale, store, per-row partial sum of exp(out-32). ---
    const float CLIP = 1.0f - 1e-6f;
#pragma unroll
    for (int mt = 0; mt < 2; ++mt) {
#pragma unroll
        for (int r = 0; r < 8; ++r) {
            int m = m0 + mt * 16 + half * 8 + r;   // C/D layout: VGPR r -> M=r (lo) / r+8 (hi)
            float esum = 0.f;
#pragma unroll
            for (int nt = 0; nt < 2; ++nt) {
                float ccos = acc[mt][nt][r];
                ccos = fminf(fmaxf(ccos, -CLIP), CLIP);
                float o = 32.0f * ccos;
                outs[(size_t)m * NCLS + (n0 + nt * 16 + l15)] = o;
                esum += __expf(o - 32.0f);
            }
#pragma unroll
            for (int msk = 8; msk >= 1; msk >>= 1) esum += __shfl_xor(esum, msk, 32);
            if (l15 == 0) atomicAdd(&rowsum[m], esum);
        }
    }
}

// Kernel 3: apply margin at target entries, fix sumexp, emit target vectors + mean CE loss.
__global__ void __launch_bounds__(512) finalize(const int* __restrict__ labels,
                                                float* __restrict__ d_out,
                                                const float* __restrict__ rowsum) {
    int i = threadIdx.x;                      // 512 threads, one per sample
    float* outs = d_out + 1;
    float* tgt  = d_out + 1 + (size_t)BATCH * NCLS;

    int lab = labels[i];
    size_t idx = (size_t)i * NCLS + (size_t)lab;
    float oc    = outs[idx];                  // 32 * clipped cos_in
    float cosin = oc * (1.0f / 32.0f);
    float theta = acosf(cosin);
    float thout = theta + 0.5f;               // M2 = 0.5, M1 = 1, M3 = 0
    const float MMc = 0.23971276930210156f;   // 0.5 * sin(pi - 0.5)
    float cosout = (thout < 3.14159265358979323846f) ? cosf(thout) : (cosin - MMc);
    float newo = 32.0f * cosout;

    outs[idx]          = newo;
    tgt[i]             = cosin;               // target_cos_in
    tgt[BATCH + i]     = theta;               // target_theta_in
    tgt[2 * BATCH + i] = cosout;              // target_cos_out

    float se    = rowsum[i] + (__expf(newo - 32.0f) - __expf(oc - 32.0f));
    float lossi = (32.0f + logf(se)) - newo;  // logsumexp - target logit

#pragma unroll
    for (int m = 16; m >= 1; m >>= 1) lossi += __shfl_xor(lossi, m, 32);
    __shared__ float red[16];
    int wave = i >> 5, lane = i & 31;
    if (lane == 0) red[wave] = lossi;
    __syncthreads();
    if (i == 0) {
        float tot = 0.f;
#pragma unroll
        for (int w = 0; w < 16; ++w) tot += red[w];
        d_out[0] = tot / (float)BATCH;
    }
}

extern "C" void kernel_launch(void* const* d_in, const int* in_sizes, int n_in,
                              void* d_out, int out_size, void* d_ws, size_t ws_size,
                              hipStream_t stream) {
    const float* inputs = (const float*)d_in[0];
    const int*   labels = (const int*)d_in[1];
    const float* weight = (const float*)d_in[2];
    float* out = (float*)d_out;

    unsigned short* Abf    = (unsigned short*)d_ws;                         // 512 KB
    float*          rowsum = (float*)((char*)d_ws + (size_t)BATCH * FEAT * 2);

    prep_inputs<<<BATCH, 256, 0, stream>>>(inputs, Abf, rowsum);
    margin_gemm<<<NCLS / NTILE, 512, 0, stream>>>(Abf, weight, out + 1, rowsum);
    finalize<<<1, BATCH, 0, stream>>>(labels, out, rowsum);
}
